// LARNN_27058293965416
// MI455X (gfx1250) — matrix operand order repaired and checked by reading, not compiled, not run on hardware
//
#include <hip/hip_runtime.h>
#include <hip/hip_bf16.h>
#include <math.h>

// ---------------------------------------------------------------------------
// LARNN on MI455X (gfx1250, wave32, WMMA).
// Sizes: T=128, B=256, F=128, H=512, W=16, NH=8, DK=64.
// All GEMMs run on v_wmma_f32_16x16x32_bf16 (bf16 A/B, f32 accum).
// K/V over the attention buffer are cached in a 16-slot ring (computed once
// per new cell state) instead of recomputed over the whole buffer each step.
// A-tiles are streamed into LDS with GLOBAL_LOAD_ASYNC_TO_LDS_B128 (ASYNCcnt)
// using double buffering so staging overlaps the WMMA work.
// ---------------------------------------------------------------------------

typedef __bf16 bf16;
typedef __attribute__((ext_vector_type(16))) __bf16 v16bf;
typedef __attribute__((ext_vector_type(8)))  float  v8f;

#define TT 128
#define BB 256
#define FF 128
#define HH 512
#define WIN 16
#define NHEAD 8
#define DKK 64

union FragU { v16bf v; uint4 q[2]; };

__device__ __forceinline__ float elu1(float x) {
  return x > 0.f ? x : (expf(x) - 1.f);
}
__device__ __forceinline__ float sigm(float x) {
  return 1.f / (1.f + expf(-x));
}

// --------- CDNA5 async global->LDS copy (ASYNCcnt) with safe fallback ------
#if defined(__AMDGCN__) && __has_builtin(__builtin_amdgcn_global_load_async_to_lds_b128)
#define HAVE_ASYNC_LDS 1
typedef int v4i_ls __attribute__((vector_size(16)));         // b128 payload
typedef __attribute__((address_space(3))) v4i_ls* lds_v4i_p; // LDS dest
#endif

__device__ __forceinline__ void async_copy16(const bf16* g, bf16* l) {
#ifdef HAVE_ASYNC_LDS
  __builtin_amdgcn_global_load_async_to_lds_b128((v4i_ls*)(void*)g,
                                                 (lds_v4i_p)l, 0, 0);
#else
  *(uint4*)l = *(const uint4*)g;
#endif
}
__device__ __forceinline__ void async_wait() {
#ifdef HAVE_ASYNC_LDS
  asm volatile("s_wait_asynccnt 0x0" ::: "memory");
#endif
}

// ----------------------------- WMMA GEMM -----------------------------------
// C[M,N] = act(A[M,K] @ W[K,N] + bias), A bf16 row-major, W supplied
// TRANSPOSED as Wt[N,K] bf16 so B-fragments are contiguous 16B runs.
// Block: 256 threads = 8 waves (2 M-waves x 4 N-waves), block tile 64x256,
// wave tile 32x64 (2x4 fragments of 16x16). LDS double-buffers the 64x32
// A tile; the next tile streams in asynchronously during compute.
#define BM 64
#define BN 256
#define BKC 32
#define LDSK 40   // padded row (byte stride 80, multiple of 16)

__device__ __forceinline__ v16bf load_frag_a(const bf16* tile, int row) {
  int lane = threadIdx.x & 31;
  int r  = row + (lane & 15);
  int hi = lane >> 4;
  FragU f;
  f.q[0] = *(const uint4*)(tile + r * LDSK + hi * 8);
  f.q[1] = *(const uint4*)(tile + r * LDSK + 16 + hi * 8);
  return f.v;
}

__device__ __forceinline__ v16bf load_frag_b(const bf16* __restrict__ Wt,
                                             int n, int ldw, int kc) {
  int lane = threadIdx.x & 31;
  int col = n + (lane & 15);
  int hi  = lane >> 4;
  const bf16* base = Wt + (size_t)col * ldw + kc + hi * 8;
  FragU f;
  f.q[0] = *(const uint4*)(base);
  f.q[1] = *(const uint4*)(base + 16);
  return f.v;
}

__global__ __launch_bounds__(256)
void gemm_bf16_wmma(const bf16* __restrict__ A, const bf16* __restrict__ Wt,
                    const float* __restrict__ bias,
                    float* outF, bf16* outB,
                    int M, int N, int K, int act) {
  __shared__ bf16 Atile[2][BM * LDSK];
  int tid  = threadIdx.x;
  int wid  = tid >> 5, lane = tid & 31;
  int wm   = wid >> 2, wn = wid & 3;
  int m0   = blockIdx.y * BM;
  int n0   = blockIdx.x * BN;

  v8f zero = {0.f, 0.f, 0.f, 0.f, 0.f, 0.f, 0.f, 0.f};
  v8f acc[2][4];
  for (int i = 0; i < 2; ++i)
    for (int j = 0; j < 4; ++j) acc[i][j] = zero;

  // per-thread staging slot: 64 rows x 32 k, one b128 per thread per tile
  int crow = tid >> 2;
  int ckp  = (tid & 3) * 8;
  const bf16* Abase = A + (size_t)(m0 + crow) * K + ckp;
  bf16* lslot[2] = { &Atile[0][crow * LDSK + ckp], &Atile[1][crow * LDSK + ckp] };

  // prologue: stage tile 0
  async_copy16(Abase, lslot[0]);
  async_wait();
  __syncthreads();

  int nk = K / BKC;
  for (int ki = 0; ki < nk; ++ki) {
    int kc  = ki * BKC;
    int cur = ki & 1;
    const bf16* tile = Atile[cur];

    // stream next A tile into the other LDS buffer while we compute
    if (ki + 1 < nk) {
      async_copy16(Abase + (kc + BKC), lslot[cur ^ 1]);
      __builtin_prefetch(Wt + (size_t)(n0 + wn * 64 + lane) * K + kc + BKC, 0, 1);
    }

    v16bf af[2];
    af[0] = load_frag_a(tile, wm * 32);
    af[1] = load_frag_a(tile, wm * 32 + 16);

    for (int nf = 0; nf < 4; ++nf) {
      v16bf bf = load_frag_b(Wt, n0 + wn * 64 + nf * 16, K, kc);
      acc[0][nf] = __builtin_amdgcn_wmma_f32_16x16x32_bf16(
          false, af[0], false, bf, (short)0, acc[0][nf], false, false);
      acc[1][nf] = __builtin_amdgcn_wmma_f32_16x16x32_bf16(
          false, af[1], false, bf, (short)0, acc[1][nf], false, false);
    }

    async_wait();        // next tile resident in LDS
    __syncthreads();     // all waves done reading current tile
  }

  // epilogue: D layout (ISA 7.12.2): lane -> N=lane&15, VGPR v -> M=v+8*(lane>=16)
  int hi = lane >> 4, nl = lane & 15;
  for (int mf = 0; mf < 2; ++mf)
    for (int nf = 0; nf < 4; ++nf)
      for (int v = 0; v < 8; ++v) {
        int gr = m0 + wm * 32 + mf * 16 + hi * 8 + v;
        int gc = n0 + wn * 64 + nf * 16 + nl;
        float val = acc[mf][nf][v] + (bias ? bias[gc] : 0.f);
        if (act == 1) val = elu1(val);
        size_t idx = (size_t)gr * N + gc;
        if (outF) outF[idx] = val;
        if (outB) outB[idx] = (bf16)val;
      }
}

// --------------------------- attention -------------------------------------
// One wave per (batch, head). Lane owns 2 of DK=64 dims. Scores via shuffle
// allreduce; masked softmax realized by iterating only valid ring entries
// (plus the zero-row K/V that the reference's initial mask keeps alive for
// the first 16 steps).
__global__ __launch_bounds__(256)
void attn_kernel(const float* __restrict__ Qh,
                 const float* __restrict__ Kring, const float* __restrict__ Vring,
                 const float* __restrict__ Kz, const float* __restrict__ Vz,
                 bf16* __restrict__ ctxb, int t) {
  int wid = threadIdx.x >> 5, lane = threadIdx.x & 31;
  int p = blockIdx.x * 8 + wid;           // 0 .. B*NH-1
  int b = p >> 3, h = p & 7;
  int d0 = lane * 2;
  const float* qp = Qh + (size_t)b * HH + h * DKK;
  float q0 = qp[d0], q1 = qp[d0 + 1];

  int nreal = t < WIN ? t : WIN;
  int hasZ  = (t < WIN) ? 1 : 0;
  int nw = nreal + hasZ;

  float s[WIN + 1];
  float mx = -1e30f;
  const float scale = 0.125f;             // 1/sqrt(64)
  for (int i = 0; i < nw; ++i) {
    const float* kp;
    if (i < nreal) {
      int j = t - nreal + i;
      kp = Kring + (size_t)(j & (WIN - 1)) * BB * HH + (size_t)b * HH + h * DKK;
    } else {
      kp = Kz + h * DKK;
    }
    float part = q0 * kp[d0] + q1 * kp[d0 + 1];
    for (int off = 16; off; off >>= 1) part += __shfl_xor(part, off, 32);
    s[i] = part * scale;
    mx = fmaxf(mx, s[i]);
  }
  float denom = 0.f;
  for (int i = 0; i < nw; ++i) { s[i] = expf(s[i] - mx); denom += s[i]; }
  float inv = 1.f / denom;

  float c0 = 0.f, c1 = 0.f;
  for (int i = 0; i < nw; ++i) {
    const float* vp;
    if (i < nreal) {
      int j = t - nreal + i;
      vp = Vring + (size_t)(j & (WIN - 1)) * BB * HH + (size_t)b * HH + h * DKK;
    } else {
      vp = Vz + h * DKK;
    }
    c0 += s[i] * vp[d0];
    c1 += s[i] * vp[d0 + 1];
  }
  bf16* op = ctxb + (size_t)b * HH + h * DKK;
  op[d0]     = (bf16)(c0 * inv);
  op[d0 + 1] = (bf16)(c1 * inv);
}

// --------------------------- batch-norm ------------------------------------
// One block per feature (H blocks), 256 threads = batch. Population variance
// (ddof=0) matching jnp.var.
__global__ __launch_bounds__(256)
void bn_kernel(const float* __restrict__ src, const float* __restrict__ gamma,
               const float* __restrict__ beta,
               float* outF, bf16* outB, float* outF2) {
  __shared__ float red[256];
  __shared__ float mean_s, var_s;
  int f = blockIdx.x;
  int b = threadIdx.x;
  float x = src[(size_t)b * HH + f];
  red[b] = x; __syncthreads();
  for (int s = 128; s; s >>= 1) { if (b < s) red[b] += red[b + s]; __syncthreads(); }
  if (b == 0) mean_s = red[0] * (1.f / BB);
  __syncthreads();
  float d = x - mean_s;
  red[b] = d * d; __syncthreads();
  for (int s = 128; s; s >>= 1) { if (b < s) red[b] += red[b + s]; __syncthreads(); }
  if (b == 0) var_s = red[0] * (1.f / BB);
  __syncthreads();
  float y = d * rsqrtf(var_s + 1e-5f) * gamma[f] + beta[f];
  size_t idx = (size_t)b * HH + f;
  if (outF)  outF[idx]  = y;
  if (outB)  outB[idx]  = (bf16)y;
  if (outF2) outF2[idx] = y;
}

// --------------------------- cell update -----------------------------------
__global__ __launch_bounds__(256)
void cell_kernel(const float* __restrict__ pre, float* __restrict__ c,
                 bf16* __restrict__ cb, float* __restrict__ tmp2) {
  int i = blockIdx.x * 256 + threadIdx.x;   // B*H
  int b = i >> 9, f = i & (HH - 1);
  const float* pr = pre + (size_t)b * (4 * HH);
  float iv = tanhf(pr[f]);
  float fg = sigm(pr[HH + f]);
  float ig = sigm(pr[2 * HH + f]);
  float og = sigm(pr[3 * HH + f]);
  float cn = iv * ig + c[i] * fg;
  c[i]  = cn;
  cb[i] = (bf16)cn;
  tmp2[i] = og * elu1(cn);
}

// --------------------------- concat / convert -------------------------------
__global__ void concat_xh(const bf16* __restrict__ xt, const bf16* __restrict__ hb,
                          bf16* __restrict__ a1) {
  int i = blockIdx.x * 256 + threadIdx.x;   // B*640
  int b = i / 640, k = i % 640;
  a1[i] = (k < FF) ? xt[b * FF + k] : hb[(size_t)b * HH + (k - FF)];
}

__global__ void concat_xha(const bf16* __restrict__ xt, const bf16* __restrict__ hb,
                           const bf16* __restrict__ attb, bf16* __restrict__ acat) {
  int i = blockIdx.x * 256 + threadIdx.x;   // B*1152
  int b = i / 1152, k = i % 1152;
  bf16 v;
  if (k < FF)            v = xt[b * FF + k];
  else if (k < FF + HH)  v = hb[(size_t)b * HH + (k - FF)];
  else                   v = attb[(size_t)b * HH + (k - FF - HH)];
  acat[i] = v;
}

__global__ void f32_to_bf16_k(const float* __restrict__ s, bf16* __restrict__ d, int n) {
  int i = blockIdx.x * 256 + threadIdx.x;
  if (i < n) d[i] = (bf16)s[i];
}

// Wt[n][k] = (bf16) W[k][n]
__global__ void transpose_bf16_k(const float* __restrict__ W, bf16* __restrict__ Wt,
                                 int K, int N) {
  int i = blockIdx.x * 256 + threadIdx.x;
  if (i < N * K) {
    int n = i / K, k = i % K;
    Wt[i] = (bf16)W[(size_t)k * N + n];
  }
}

// WcatT[n][k] for [W_ih(128); W_hh(512); W_ac(512)] -> K=1152, N=2048
__global__ void build_wcat_t(const float* __restrict__ Wih, const float* __restrict__ Whh,
                             const float* __restrict__ Wac, bf16* __restrict__ Wt) {
  int i = blockIdx.x * 256 + threadIdx.x;
  if (i < 2048 * 1152) {
    int n = i / 1152, k = i % 1152;
    float v;
    if (k < 128)      v = Wih[(size_t)k * 2048 + n];
    else if (k < 640) v = Whh[(size_t)(k - 128) * 2048 + n];
    else              v = Wac[(size_t)(k - 640) * 2048 + n];
    Wt[i] = (bf16)v;
  }
}

__global__ void init_state_k(const float* __restrict__ bih, const float* __restrict__ bac,
                             const float* __restrict__ bk, const float* __restrict__ bv,
                             float* bcat, float* Kz, float* Vz,
                             float* h, bf16* hb, float* c, bf16* cb) {
  int i = blockIdx.x * 256 + threadIdx.x;   // grid covers B*H
  if (i < 4 * HH) bcat[i] = bih[i] + bac[i];
  if (i < HH) { Kz[i] = elu1(bk[i]); Vz[i] = elu1(bv[i]); }
  if (i < BB * HH) {
    h[i] = 0.f; c[i] = 0.f;
    hb[i] = (bf16)0.f; cb[i] = (bf16)0.f;
  }
}

// ---------------------------------------------------------------------------
extern "C" void kernel_launch(void* const* d_in, const int* in_sizes, int n_in,
                              void* d_out, int out_size, void* d_ws, size_t ws_size,
                              hipStream_t stream) {
  const float* x     = (const float*)d_in[0];
  const float* W_ih  = (const float*)d_in[1];
  const float* b_ih  = (const float*)d_in[2];
  const float* W_hh  = (const float*)d_in[3];
  const float* Wq_in = (const float*)d_in[4];
  const float* bq_in = (const float*)d_in[5];
  const float* Wq    = (const float*)d_in[6];
  const float* bq    = (const float*)d_in[7];
  const float* Wk    = (const float*)d_in[8];
  const float* bk    = (const float*)d_in[9];
  const float* Wv    = (const float*)d_in[10];
  const float* bv    = (const float*)d_in[11];
  const float* Wo    = (const float*)d_in[12];
  const float* bo    = (const float*)d_in[13];
  const float* W_ac  = (const float*)d_in[14];
  const float* b_ac  = (const float*)d_in[15];
  const float* g_att   = (const float*)d_in[16];
  const float* be_att  = (const float*)d_in[17];
  const float* g_post  = (const float*)d_in[18];
  const float* be_post = (const float*)d_in[19];
  float* hs = (float*)d_out;

  char* ws = (char*)d_ws;
  size_t off = 0;
  auto alloc = [&](size_t bytes) -> void* {
    void* p = ws + off;
    off += (bytes + 255) & ~(size_t)255;
    return p;
  };

  bf16* xb     = (bf16*)alloc((size_t)TT * BB * FF * 2);
  bf16* WqinT  = (bf16*)alloc((size_t)HH * 640 * 2);
  bf16* WqT    = (bf16*)alloc((size_t)HH * HH * 2);
  bf16* WkT    = (bf16*)alloc((size_t)HH * HH * 2);
  bf16* WvT    = (bf16*)alloc((size_t)HH * HH * 2);
  bf16* WoT    = (bf16*)alloc((size_t)HH * HH * 2);
  bf16* WcatT  = (bf16*)alloc((size_t)2048 * 1152 * 2);
  float* bcat  = (float*)alloc(2048 * 4);
  float* Kz    = (float*)alloc(HH * 4);
  float* Vz    = (float*)alloc(HH * 4);
  float* h     = (float*)alloc((size_t)BB * HH * 4);
  bf16*  hb    = (bf16*) alloc((size_t)BB * HH * 2);
  float* c     = (float*)alloc((size_t)BB * HH * 4);
  bf16*  cb    = (bf16*) alloc((size_t)BB * HH * 2);
  bf16*  qb    = (bf16*) alloc((size_t)BB * HH * 2);
  float* Qh    = (float*)alloc((size_t)BB * HH * 4);
  bf16*  a1    = (bf16*) alloc((size_t)BB * 640 * 2);
  bf16*  acat  = (bf16*) alloc((size_t)BB * 1152 * 2);
  bf16*  ctxb  = (bf16*) alloc((size_t)BB * HH * 2);
  float* attpre= (float*)alloc((size_t)BB * HH * 4);
  bf16*  attb  = (bf16*) alloc((size_t)BB * HH * 2);
  float* pre   = (float*)alloc((size_t)BB * 4 * HH * 4);
  float* tmp2  = (float*)alloc((size_t)BB * HH * 4);
  float* Kring = (float*)alloc((size_t)WIN * BB * HH * 4);
  float* Vring = (float*)alloc((size_t)WIN * BB * HH * 4);

  // ---- one-time setup (deterministic, re-run every call) ----
  {
    int n = TT * BB * FF;
    f32_to_bf16_k<<<(n + 255) / 256, 256, 0, stream>>>(x, xb, n);
    transpose_bf16_k<<<(640 * HH + 255) / 256, 256, 0, stream>>>(Wq_in, WqinT, 640, HH);
    transpose_bf16_k<<<(HH * HH + 255) / 256, 256, 0, stream>>>(Wq, WqT, HH, HH);
    transpose_bf16_k<<<(HH * HH + 255) / 256, 256, 0, stream>>>(Wk, WkT, HH, HH);
    transpose_bf16_k<<<(HH * HH + 255) / 256, 256, 0, stream>>>(Wv, WvT, HH, HH);
    transpose_bf16_k<<<(HH * HH + 255) / 256, 256, 0, stream>>>(Wo, WoT, HH, HH);
    build_wcat_t<<<(2048 * 1152 + 255) / 256, 256, 0, stream>>>(W_ih, W_hh, W_ac, WcatT);
    init_state_k<<<(BB * HH + 255) / 256, 256, 0, stream>>>(
        b_ih, b_ac, bk, bv, bcat, Kz, Vz, h, hb, c, cb);
  }

  dim3 blk(256);
  dim3 g512(HH / BN, BB / BM);     // N=512  -> (2,4)
  dim3 g2048(2048 / BN, BB / BM);  // N=2048 -> (8,4)

  for (int t = 0; t < TT; ++t) {
    const bf16* xt = xb + (size_t)t * BB * FF;

    // q = elu([x_t, h] @ Wq_in + bq_in)   (kept in bf16 for next GEMM)
    concat_xh<<<(BB * 640) / 256, blk, 0, stream>>>(xt, hb, a1);
    gemm_bf16_wmma<<<g512, blk, 0, stream>>>(a1, WqinT, bq_in, nullptr, qb,
                                             BB, HH, 640, 1);
    // Qh = q @ Wq + bq
    gemm_bf16_wmma<<<g512, blk, 0, stream>>>(qb, WqT, bq, Qh, nullptr,
                                             BB, HH, HH, 0);
    // windowed attention over cached K/V ring
    attn_kernel<<<(BB * NHEAD) / 8, blk, 0, stream>>>(Qh, Kring, Vring, Kz, Vz,
                                                      ctxb, t);
    // att = BN(elu(ctx @ Wo + bo))
    gemm_bf16_wmma<<<g512, blk, 0, stream>>>(ctxb, WoT, bo, attpre, nullptr,
                                             BB, HH, HH, 1);
    bn_kernel<<<HH, blk, 0, stream>>>(attpre, g_att, be_att, nullptr, attb, nullptr);

    // pre = [x_t, h, att] @ [W_ih; W_hh; W_ac] + (b_ih + b_ac)
    concat_xha<<<(BB * 1152) / 256, blk, 0, stream>>>(xt, hb, attb, acat);
    gemm_bf16_wmma<<<g2048, blk, 0, stream>>>(acat, WcatT, bcat, pre, nullptr,
                                              BB, 4 * HH, 1152, 0);
    // gates, cell state, post-activation
    cell_kernel<<<(BB * HH) / 256, blk, 0, stream>>>(pre, c, cb, tmp2);
    // h_new = BN(o_g * elu(c_new)); also emit hs[t]
    bn_kernel<<<HH, blk, 0, stream>>>(tmp2, g_post, be_post, h, hb,
                                      hs + (size_t)t * BB * HH);
    // append K/V for c_t into ring slot (used from step t+1 on)
    float* kslot = Kring + (size_t)(t & (WIN - 1)) * BB * HH;
    float* vslot = Vring + (size_t)(t & (WIN - 1)) * BB * HH;
    gemm_bf16_wmma<<<g512, blk, 0, stream>>>(cb, WkT, bk, kslot, nullptr,
                                             BB, HH, HH, 1);
    gemm_bf16_wmma<<<g512, blk, 0, stream>>>(cb, WvT, bv, vslot, nullptr,
                                             BB, HH, HH, 1);
  }
}
